// TwoWayCrossAttentionSingleInput_22265110462906
// MI455X (gfx1250) — compile-verified, matrix operand order checked by reading, not run on hardware
//
#include <hip/hip_runtime.h>
#include <hip/hip_bf16.h>

// ---------------- CDNA5 (gfx1250) wave32 WMMA types ----------------
typedef __bf16 bf16_t;
typedef bf16_t v16bf __attribute__((ext_vector_type(16)));
typedef bf16_t v8bf  __attribute__((ext_vector_type(8)));
typedef float  v8f   __attribute__((ext_vector_type(8)));

constexpr int kDim   = 1024;   // channel dim
constexpr int kHeads = 16;
constexpr int kHd    = 64;     // head dim
constexpr int kNq    = 256;    // num_queries
constexpr int kNc    = 4096;   // context tokens (4352-256)
constexpr int kNb    = 4;      // batch
constexpr int kRows  = 4352;   // tokens per batch in x

// Load a 16x32 bf16 tile in the wave32 WMMA operand layout.
// For the A operand: src is row-major (row m, k contiguous), r = m.
// For the B operand: src is the *transposed* matrix (row n, k contiguous), r = n.
// Per ISA 7.12.2: lanes 0-15 hold k 0..7 / 16..23, lanes 16-31 hold k 8..15 / 24..31.
__device__ __forceinline__ v16bf load_tile16x32(const bf16_t* __restrict__ src, int ld) {
  const int lane = threadIdx.x & 31;
  const int r    = lane & 15;
  const int sel  = lane >> 4;
  const bf16_t* p = src + (size_t)r * ld + sel * 8;
  union { v16bf v; v8bf h[2]; } u;
  u.h[0] = *(const v8bf*)(p);        // global_load_b128
  u.h[1] = *(const v8bf*)(p + 16);   // global_load_b128
  return u.v;
}

__device__ __forceinline__ v8f wmma_bf16(v16bf a, v16bf b, v8f c) {
  // D(f32 16x16) = A(bf16 16x32) * B(bf16 32x16) + C
  return __builtin_amdgcn_wmma_f32_16x16x32_bf16(false, a, false, b, (short)0, c,
                                                 false, false);
}

// ---------------- input staging ----------------
__global__ __launch_bounds__(256) void split_convert_x(
    const float* __restrict__ x, bf16_t* __restrict__ xq, bf16_t* __restrict__ xc) {
  const size_t idx = (size_t)blockIdx.x * blockDim.x + threadIdx.x;
  const int    col = (int)(idx & (kDim - 1));
  const size_t row = idx >> 10;
  const int b = (int)(row / kRows);
  const int r = (int)(row % kRows);
  const bf16_t v = (bf16_t)x[idx];
  if (r < kNq) xq[((size_t)b * kNq + r) * kDim + col] = v;
  else         xc[((size_t)b * kNc + (r - kNq)) * kDim + col] = v;
}

// W (K x N fp32, row-major) -> WT (N x K bf16, row-major) via LDS tile transpose.
__global__ __launch_bounds__(256) void transpose_w_bf16(
    const float* __restrict__ W, bf16_t* __restrict__ WT, int K, int N) {
  __shared__ bf16_t t[32][33];
  const int n0 = blockIdx.x * 32, k0 = blockIdx.y * 32;
  const int tx = threadIdx.x & 31, ty = threadIdx.x >> 5;
  #pragma unroll
  for (int i = ty; i < 32; i += 8)
    t[i][tx] = (bf16_t)W[(size_t)(k0 + i) * N + (n0 + tx)];
  __syncthreads();
  #pragma unroll
  for (int i = ty; i < 32; i += 8)
    WT[(size_t)(n0 + i) * K + (k0 + tx)] = t[tx][i];
}

// ---------------- generic WMMA GEMM with fused epilogues ----------------
// C[M x N] = A[M x K](bf16 rm) * WT[N x K](bf16, row n holds column n of W)
// Each wave: 32x64 output (2x4 tiles of 16x16), 8 v_wmma per K=32 step.
// mode 0: Q proj   -> outA[((b*H+h)*rowsPerB + r)*64 + dd]           (bf16)
// mode 1: KV proj  -> K to outA (row-major per bh), V to outB transposed (64 x Nk)
// mode 2: out proj -> outF = acc + bias + resid (fp32, x-layout); outA = bf16 copy
__global__ __launch_bounds__(256) void gemm_bf16_wmma(
    const bf16_t* __restrict__ A, const bf16_t* __restrict__ WT,
    int M, int N, int K, int mode, int rowsPerB, int rowOff,
    bf16_t* __restrict__ outA, bf16_t* __restrict__ outB,
    const float* __restrict__ bias, const float* __restrict__ resid,
    float* __restrict__ outF) {
  const int lane = threadIdx.x & 31;
  const int wave = threadIdx.x >> 5;
  const int m0   = (blockIdx.y * 8 + wave) * 32;
  const int n0   = blockIdx.x * 64;
  if (m0 >= M) return;

  v8f acc[2][4] = {};
  for (int k0 = 0; k0 < K; k0 += 32) {
    v16bf a0 = load_tile16x32(A + (size_t)m0 * K + k0, K);
    v16bf a1 = load_tile16x32(A + (size_t)(m0 + 16) * K + k0, K);
    #pragma unroll
    for (int j = 0; j < 4; ++j) {
      v16bf bt = load_tile16x32(WT + (size_t)(n0 + j * 16) * K + k0, K);
      acc[0][j] = wmma_bf16(a0, bt, acc[0][j]);
      acc[1][j] = wmma_bf16(a1, bt, acc[1][j]);
    }
  }

  const int sel = lane >> 4, ln = lane & 15;
  const int bIdx  = m0 / rowsPerB;           // wave tile never crosses a batch
  const int rBase = m0 % rowsPerB;
  #pragma unroll
  for (int i = 0; i < 2; ++i) {
    #pragma unroll
    for (int j = 0; j < 4; ++j) {
      const int gn = n0 + j * 16 + ln;
      #pragma unroll
      for (int v = 0; v < 8; ++v) {
        const int   r   = rBase + i * 16 + sel * 8 + v;
        const float val = acc[i][j][v];
        if (mode == 0) {
          const int h = gn >> 6, dd = gn & 63;
          outA[(((size_t)(bIdx * kHeads + h)) * rowsPerB + r) * kHd + dd] = (bf16_t)val;
        } else if (mode == 1) {
          const int s = gn >> 10, h = (gn >> 6) & 15, dd = gn & 63;
          if (s == 0)
            outA[(((size_t)(bIdx * kHeads + h)) * rowsPerB + r) * kHd + dd] = (bf16_t)val;
          else
            outB[(((size_t)(bIdx * kHeads + h)) * kHd + dd) * rowsPerB + r] = (bf16_t)val;
        } else {
          const size_t o = ((size_t)bIdx * kRows + rowOff + r) * kDim + gn;
          const float  y = val + bias[gn] + resid[o];
          outF[o] = y;
          if (outA) outA[((size_t)(bIdx * rowsPerB) + r) * kDim + gn] = (bf16_t)y;
        }
      }
    }
  }
}

// ---------------- flash attention, one wave per 16 query rows ----------------
__global__ __launch_bounds__(256) void flash_attn_wmma(
    const bf16_t* __restrict__ Qb, const bf16_t* __restrict__ Kb,
    const bf16_t* __restrict__ VTb, bf16_t* __restrict__ Ob,
    int Nq, int Nk, int groups) {
  __shared__ __align__(16) bf16_t pshare[8 * 16 * 32];  // 1KB per wave: P transpose
  const int lane = threadIdx.x & 31;
  const int wave = threadIdx.x >> 5;
  bf16_t* pl = pshare + wave * 512;

  const int bh  = blockIdx.x / groups;
  const int grp = blockIdx.x % groups;
  const int qt  = grp * 8 + wave;
  const int b   = bh >> 4, h = bh & 15;
  const int sel = lane >> 4, ln = lane & 15;

  const bf16_t* Q  = Qb  + ((size_t)bh * Nq + qt * 16) * kHd;
  const bf16_t* Kp = Kb  + (size_t)bh * Nk * kHd;
  const bf16_t* VT = VTb + (size_t)bh * kHd * Nk;

  const v16bf qa0 = load_tile16x32(Q, kHd);        // k-dim 0..31 of d=64
  const v16bf qa1 = load_tile16x32(Q + 32, kHd);   // k-dim 32..63
  v8f o0 = {}, o1 = {}, o2 = {}, o3 = {};
  float mr[8], lr[8];
  #pragma unroll
  for (int v = 0; v < 8; ++v) { mr[v] = -1e30f; lr[v] = 0.0f; }
  const float scale = 0.125f;  // 1/sqrt(64)

  for (int kc = 0; kc < Nk; kc += 32) {
    const bf16_t* Kc = Kp + (size_t)kc * kHd;
    // S(16q x 32k) = Q(16x64) * K^T : K rows row-major == "transposed B" layout
    v8f s0 = {}, s1 = {};
    {
      v16bf kb0 = load_tile16x32(Kc, kHd);
      v16bf kb1 = load_tile16x32(Kc + 16 * kHd, kHd);
      s0 = wmma_bf16(qa0, kb0, s0);
      s1 = wmma_bf16(qa0, kb1, s1);
      kb0 = load_tile16x32(Kc + 32, kHd);
      kb1 = load_tile16x32(Kc + 16 * kHd + 32, kHd);
      s0 = wmma_bf16(qa1, kb0, s0);
      s1 = wmma_bf16(qa1, kb1, s1);
    }
    if (kc + 32 < Nk) __builtin_prefetch(Kc + 32 * kHd, 0, 3);  // global_prefetch

    float mnew[8], corr[8], rs[8], p0[8], p1[8];
    #pragma unroll
    for (int v = 0; v < 8; ++v) { s0[v] *= scale; s1[v] *= scale; }
    #pragma unroll
    for (int v = 0; v < 8; ++v) mnew[v] = fmaxf(mr[v], fmaxf(s0[v], s1[v]));
    #pragma unroll
    for (int off = 1; off < 16; off <<= 1) {   // reduce within 16-lane C-layout half
      #pragma unroll
      for (int v = 0; v < 8; ++v) mnew[v] = fmaxf(mnew[v], __shfl_xor(mnew[v], off, 32));
    }
    #pragma unroll
    for (int v = 0; v < 8; ++v) {
      corr[v] = __expf(mr[v] - mnew[v]);
      mr[v]   = mnew[v];
      p0[v]   = __expf(s0[v] - mnew[v]);
      p1[v]   = __expf(s1[v] - mnew[v]);
      rs[v]   = p0[v] + p1[v];
    }
    #pragma unroll
    for (int off = 1; off < 16; off <<= 1) {
      #pragma unroll
      for (int v = 0; v < 8; ++v) rs[v] += __shfl_xor(rs[v], off, 32);
    }
    #pragma unroll
    for (int v = 0; v < 8; ++v) {
      lr[v] = lr[v] * corr[v] + rs[v];
      o0[v] *= corr[v]; o1[v] *= corr[v]; o2[v] *= corr[v]; o3[v] *= corr[v];
      // stage P (C-layout -> row-major) in wave-private LDS; DS is in-order per wave
      pl[(sel * 8 + v) * 32 + ln]      = (bf16_t)p0[v];
      pl[(sel * 8 + v) * 32 + 16 + ln] = (bf16_t)p1[v];
    }
    v16bf pa;
    {
      const bf16_t* pp = pl + ln * 32 + sel * 8;   // reload in A-operand layout
      union { v16bf v; v8bf h2[2]; } u;
      u.h2[0] = *(const v8bf*)(pp);
      u.h2[1] = *(const v8bf*)(pp + 16);
      pa = u.v;
    }
    // O(16q x 64d) += P(16x32) * V : VT rows (d-major) == "transposed B" layout
    const bf16_t* Vc = VT + kc;
    o0 = wmma_bf16(pa, load_tile16x32(Vc,                     Nk), o0);
    o1 = wmma_bf16(pa, load_tile16x32(Vc + 16 * (size_t)Nk,   Nk), o1);
    o2 = wmma_bf16(pa, load_tile16x32(Vc + 32 * (size_t)Nk,   Nk), o2);
    o3 = wmma_bf16(pa, load_tile16x32(Vc + 48 * (size_t)Nk,   Nk), o3);
  }

  #pragma unroll
  for (int v = 0; v < 8; ++v) {
    const float  inv = 1.0f / lr[v];
    const size_t row = ((size_t)(b * Nq) + qt * 16 + sel * 8 + v) * kDim + h * kHd + ln;
    Ob[row     ] = (bf16_t)(o0[v] * inv);
    Ob[row + 16] = (bf16_t)(o1[v] * inv);
    Ob[row + 32] = (bf16_t)(o2[v] * inv);
    Ob[row + 48] = (bf16_t)(o3[v] * inv);
  }
}

// ---------------- host-side orchestration ----------------
extern "C" void kernel_launch(void* const* d_in, const int* in_sizes, int n_in,
                              void* d_out, int out_size, void* d_ws, size_t ws_size,
                              hipStream_t stream) {
  const float* x     = (const float*)d_in[0];
  const float* wq1   = (const float*)d_in[1];
  const float* wkv1  = (const float*)d_in[2];
  const float* wq2   = (const float*)d_in[3];
  const float* wkv2  = (const float*)d_in[4];
  const float* wproj = (const float*)d_in[5];
  const float* bproj = (const float*)d_in[6];
  float* out = (float*)d_out;
  char*  ws  = (char*)d_ws;

  const size_t MB = 1024 * 1024;
  bf16_t* XQ    = (bf16_t*)(ws + 0);        //  2 MB  queries bf16
  bf16_t* XC    = (bf16_t*)(ws + 2 * MB);   // 32 MB  context bf16
  bf16_t* WQ1T  = (bf16_t*)(ws + 34 * MB);  //  2 MB
  bf16_t* WQ2T  = (bf16_t*)(ws + 36 * MB);  //  2 MB
  bf16_t* WPT   = (bf16_t*)(ws + 38 * MB);  //  2 MB
  bf16_t* WKV1T = (bf16_t*)(ws + 40 * MB);  //  4 MB
  bf16_t* WKV2T = (bf16_t*)(ws + 44 * MB);  //  4 MB
  bf16_t* Q1    = (bf16_t*)(ws + 48 * MB);  //  2 MB
  bf16_t* K1    = (bf16_t*)(ws + 50 * MB);  // 32 MB
  bf16_t* VT1   = (bf16_t*)(ws + 82 * MB);  // 32 MB
  bf16_t* O1    = (bf16_t*)(ws + 114 * MB); //  2 MB
  bf16_t* AQ    = (bf16_t*)(ws + 116 * MB); //  2 MB  (total 118 MB)
  bf16_t* Q2  = K1;    // phase-1 K dead before phase-2 Q is written
  bf16_t* K2  = Q1;
  bf16_t* VT2 = O1;
  bf16_t* O2  = VT1;

  // stage inputs
  {
    const size_t total = (size_t)kNb * kRows * kDim;
    split_convert_x<<<dim3((unsigned)(total / 256)), 256, 0, stream>>>(x, XQ, XC);
  }
  transpose_w_bf16<<<dim3(kDim / 32, kDim / 32), 256, 0, stream>>>(wq1,   WQ1T,  kDim, kDim);
  transpose_w_bf16<<<dim3(2 * kDim / 32, kDim / 32), 256, 0, stream>>>(wkv1, WKV1T, kDim, 2 * kDim);
  transpose_w_bf16<<<dim3(kDim / 32, kDim / 32), 256, 0, stream>>>(wq2,   WQ2T,  kDim, kDim);
  transpose_w_bf16<<<dim3(2 * kDim / 32, kDim / 32), 256, 0, stream>>>(wkv2, WKV2T, kDim, 2 * kDim);
  transpose_w_bf16<<<dim3(kDim / 32, kDim / 32), 256, 0, stream>>>(wproj, WPT,   kDim, kDim);

  // ---- Phase 1: queries attend to context ----
  gemm_bf16_wmma<<<dim3(kDim / 64, (kNb * kNq) / 256), 256, 0, stream>>>(
      XQ, WQ1T, kNb * kNq, kDim, kDim, 0, kNq, 0, Q1, nullptr, nullptr, nullptr, nullptr);
  gemm_bf16_wmma<<<dim3(2 * kDim / 64, (kNb * kNc) / 256), 256, 0, stream>>>(
      XC, WKV1T, kNb * kNc, 2 * kDim, kDim, 1, kNc, 0, K1, VT1, nullptr, nullptr, nullptr);
  flash_attn_wmma<<<dim3(kNb * kHeads * (kNq / 128)), 256, 0, stream>>>(
      Q1, K1, VT1, O1, kNq, kNc, kNq / 128);
  gemm_bf16_wmma<<<dim3(kDim / 64, (kNb * kNq) / 256), 256, 0, stream>>>(
      O1, WPT, kNb * kNq, kDim, kDim, 2, kNq, 0, AQ, nullptr, bproj, x, out);

  // ---- Phase 2: context attends to attended queries ----
  gemm_bf16_wmma<<<dim3(kDim / 64, (kNb * kNc) / 256), 256, 0, stream>>>(
      XC, WQ2T, kNb * kNc, kDim, kDim, 0, kNc, 0, Q2, nullptr, nullptr, nullptr, nullptr);
  gemm_bf16_wmma<<<dim3(2 * kDim / 64, (kNb * kNq) / 256), 256, 0, stream>>>(
      AQ, WKV2T, kNb * kNq, 2 * kDim, kDim, 1, kNq, 0, K2, VT2, nullptr, nullptr, nullptr);
  flash_attn_wmma<<<dim3(kNb * kHeads * (kNc / 128)), 256, 0, stream>>>(
      Q2, K2, VT2, O2, kNc, kNq, kNc / 128);
  gemm_bf16_wmma<<<dim3(kDim / 64, (kNb * kNc) / 256), 256, 0, stream>>>(
      O2, WPT, kNb * kNc, kDim, kDim, 2, kNc, kNq, nullptr, nullptr, bproj, x, out);
}
